// DFlashAttention_16939351015439
// MI455X (gfx1250) — matrix operand order repaired and checked by reading, not compiled, not run on hardware
//
#include <hip/hip_runtime.h>
#include <stdint.h>

// Problem constants (B=1)
#define QL   2048
#define CTXL 2048
#define HDIM 4096
#define NH   32
#define NKV  8
#define HD   128
#define KT   (CTXL + QL)      // 4096 total keys after concat
#define QKD  (NH * HD)        // 4096
#define KVD  (NKV * HD)       // 1024

typedef __attribute__((ext_vector_type(16))) __bf16 v16bf;
typedef __attribute__((ext_vector_type(8)))  float  v8f;

// b128 payload type for the async-to-LDS builtin (matches clang's signature:
// pointee is a GCC-style int 4-vector; AS1 = global, AS3 = LDS)
typedef int v4i __attribute__((vector_size(16)));
typedef __attribute__((address_space(1))) v4i as1_v4i;
typedef __attribute__((address_space(3))) v4i as3_v4i;

// ---- CDNA5 feature probes -------------------------------------------------
#if defined(__has_builtin)
#  if __has_builtin(__builtin_amdgcn_global_load_async_to_lds_b128) && \
      __has_builtin(__builtin_amdgcn_s_wait_asynccnt)
#    define HAS_ASYNC_LDS 1
#  else
#    define HAS_ASYNC_LDS 0
#  endif
#  if __has_builtin(__builtin_amdgcn_update_dpp)
#    define HAS_DPP 1
#  else
#    define HAS_DPP 0
#  endif
#else
#  define HAS_ASYNC_LDS 0
#  define HAS_DPP 0
#endif

union FragAB {
    v16bf v;
    unsigned short u[16];
    uint4 q[2];
};

__device__ __forceinline__ unsigned short f32_to_bf16(float f) {
    union { float f; uint32_t u; } c; c.f = f;
    uint32_t u = c.u;
    uint32_t r = u + 0x7FFFu + ((u >> 16) & 1u);     // round-to-nearest-even
    if ((u & 0x7F800000u) == 0x7F800000u) r = u;     // keep inf/nan payload
    return (unsigned short)(r >> 16);
}
__device__ __forceinline__ float bf16_to_f32(unsigned short h) {
    union { uint32_t u; float f; } c; c.u = ((uint32_t)h) << 16;
    return c.f;
}

__device__ __forceinline__ void store_out(float* C, size_t i, float v) { C[i] = v; }
__device__ __forceinline__ void store_out(unsigned short* C, size_t i, float v) { C[i] = f32_to_bf16(v); }

// Async global->LDS b128 copy (VGPR-free data path, tracked by ASYNCcnt)
__device__ __forceinline__ void async_copy_b128(const unsigned short* gsrc,
                                                unsigned short* ldst) {
#if HAS_ASYNC_LDS
    __builtin_amdgcn_global_load_async_to_lds_b128((as1_v4i*)gsrc, (as3_v4i*)ldst, 0, 0);
#else
    *(uint4*)ldst = *(const uint4*)gsrc;
#endif
}
__device__ __forceinline__ void wait_async() {
#if HAS_ASYNC_LDS
    __builtin_amdgcn_s_wait_asynccnt(0);
#endif
}

// DPP16 butterfly step (stays within a 16-lane half of the wave32)
template <int CTRL>
__device__ __forceinline__ float dpp_perm_f32(float x) {
#if HAS_DPP
    int i = __builtin_bit_cast(int, x);
    i = __builtin_amdgcn_update_dpp(i, i, CTRL, 0xf, 0xf, true);
    return __builtin_bit_cast(float, i);
#else
    return x;
#endif
}
// Reduce (max / sum) across the 16 lanes of this half; result broadcast to all 16.
__device__ __forceinline__ float half16_max(float t) {
#if HAS_DPP
    t = fmaxf(t, dpp_perm_f32<0x0B1>(t));   // quad_perm [1,0,3,2]
    t = fmaxf(t, dpp_perm_f32<0x04E>(t));   // quad_perm [2,3,0,1]
    t = fmaxf(t, dpp_perm_f32<0x141>(t));   // row_half_mirror
    t = fmaxf(t, dpp_perm_f32<0x140>(t));   // row_mirror
#else
#pragma unroll
    for (int off = 8; off >= 1; off >>= 1)
        t = fmaxf(t, __shfl_xor(t, off, 32));
#endif
    return t;
}
__device__ __forceinline__ float half16_sum(float t) {
#if HAS_DPP
    t += dpp_perm_f32<0x0B1>(t);
    t += dpp_perm_f32<0x04E>(t);
    t += dpp_perm_f32<0x141>(t);
    t += dpp_perm_f32<0x140>(t);
#else
#pragma unroll
    for (int off = 8; off >= 1; off >>= 1)
        t += __shfl_xor(t, off, 32);
#endif
    return t;
}

// ---------------------------------------------------------------------------
// fp32 -> bf16 conversion (grid-stride)
// ---------------------------------------------------------------------------
__global__ void k_f32_to_bf16(const float* __restrict__ src,
                              unsigned short* __restrict__ dst, int n) {
    int i = blockIdx.x * blockDim.x + threadIdx.x;
    int stride = gridDim.x * blockDim.x;
    for (; i < n; i += stride) dst[i] = f32_to_bf16(src[i]);
}

// ---------------------------------------------------------------------------
// RoPE, in place on bf16 buffer laid out (rows, nheads*HD); cos/sin fp32 (rows,HD)
// ---------------------------------------------------------------------------
__global__ void k_rope(unsigned short* __restrict__ x,
                       const float* __restrict__ cosb,
                       const float* __restrict__ sinb,
                       int rows, int nheads) {
    int t = blockIdx.x * blockDim.x + threadIdx.x;
    int total = rows * nheads * (HD / 2);
    if (t >= total) return;
    int d = t % (HD / 2);
    int h = (t / (HD / 2)) % nheads;
    int r = t / ((HD / 2) * nheads);
    size_t base = (size_t)r * (nheads * HD) + (size_t)h * HD;
    float x1 = bf16_to_f32(x[base + d]);
    float x2 = bf16_to_f32(x[base + d + HD / 2]);
    float c1 = cosb[(size_t)r * HD + d];
    float s1 = sinb[(size_t)r * HD + d];
    float c2 = cosb[(size_t)r * HD + d + HD / 2];
    float s2 = sinb[(size_t)r * HD + d + HD / 2];
    x[base + d]          = f32_to_bf16(x1 * c1 - x2 * s1);
    x[base + d + HD / 2] = f32_to_bf16(x2 * c2 + x1 * s2);
}

// ---------------------------------------------------------------------------
// bf16 GEMM: C[M,N] = A[M,K] * B[N,K]^T    (row-major, K contiguous)
// 256 threads = 8 waves; block tile 128x128x32; wave tile 32x64 (2x4 frags).
// Double-buffered async global->LDS staging; 1 barrier per K-step.
// ---------------------------------------------------------------------------
template <typename OutT>
__global__ __launch_bounds__(256)
void k_gemm_bf16(const unsigned short* __restrict__ A,
                 const unsigned short* __restrict__ B,
                 OutT* __restrict__ C, int M, int N, int K) {
    const int BK = 32, LDSK = BK + 8;
    __shared__ __align__(16) unsigned short As[2][128 * (32 + 8)];
    __shared__ __align__(16) unsigned short Bs[2][128 * (32 + 8)];

    const int tid  = threadIdx.x;
    const int lane = tid & 31;
    const int wave = tid >> 5;
    const int hlf  = lane >> 4;
    const int lr   = lane & 15;
    const int wm   = wave & 3;        // 4 wave rows of 32
    const int wn   = wave >> 2;       // 2 wave cols of 64
    const int m0   = blockIdx.y * 128;
    const int n0   = blockIdx.x * 128;

    v8f acc[2][4];
    v8f zero = {};
#pragma unroll
    for (int i = 0; i < 2; ++i)
#pragma unroll
        for (int j = 0; j < 4; ++j) acc[i][j] = zero;

    const int lrow = tid >> 1;
    const int lseg = (tid & 1) * 16;

    auto stage = [&](int buf, int k0) {
        const unsigned short* ga = &A[(size_t)(m0 + lrow) * K + k0 + lseg];
        const unsigned short* gb = &B[(size_t)(n0 + lrow) * K + k0 + lseg];
        unsigned short* la = &As[buf][lrow * LDSK + lseg];
        unsigned short* lb = &Bs[buf][lrow * LDSK + lseg];
        async_copy_b128(ga,     la);
        async_copy_b128(ga + 8, la + 8);
        async_copy_b128(gb,     lb);
        async_copy_b128(gb + 8, lb + 8);
    };

    stage(0, 0);
    wait_async();
    __syncthreads();

    int buf = 0;
    for (int k0 = 0; k0 < K; k0 += BK) {
        const bool more = (k0 + BK) < K;
        if (more) stage(buf ^ 1, k0 + BK);   // async fill of other buffer

        FragAB afr[2], bfr[4];
#pragma unroll
        for (int i = 0; i < 2; ++i) {        // A frag: M=lr, K chunks (hlf*8),(16+hlf*8)
            int m = wm * 32 + i * 16 + lr;
            afr[i].q[0] = *(const uint4*)&As[buf][m * LDSK + hlf * 8];
            afr[i].q[1] = *(const uint4*)&As[buf][m * LDSK + 16 + hlf * 8];
        }
#pragma unroll
        for (int j = 0; j < 4; ++j) {        // B frag: N=lr, K chunks (hlf*8),(16+hlf*8)
            int n = wn * 64 + j * 16 + lr;
            bfr[j].q[0] = *(const uint4*)&Bs[buf][n * LDSK + hlf * 8];
            bfr[j].q[1] = *(const uint4*)&Bs[buf][n * LDSK + 16 + hlf * 8];
        }
#pragma unroll
        for (int i = 0; i < 2; ++i)
#pragma unroll
            for (int j = 0; j < 4; ++j)
                acc[i][j] = __builtin_amdgcn_wmma_f32_16x16x32_bf16(
                    false, afr[i].v, false, bfr[j].v, (short)0, acc[i][j], false, false);

        if (more) {
            wait_async();                    // next buffer resident in LDS
            __syncthreads();
            buf ^= 1;
        }
    }

#pragma unroll
    for (int i = 0; i < 2; ++i) {
        int mb = m0 + wm * 32 + i * 16 + hlf * 8;
#pragma unroll
        for (int j = 0; j < 4; ++j) {
            int nb = n0 + wn * 64 + j * 16 + lr;
#pragma unroll
            for (int r = 0; r < 8; ++r)
                store_out(C, (size_t)(mb + r) * N + nb, acc[i][j][r]);
        }
    }
}

// ---------------------------------------------------------------------------
// Flash attention (full, non-causal). 128 threads = 4 waves; each wave owns
// 16 q-rows of a 64-row block. Key blocks of 32. Online softmax in fp32,
// reductions via DPP16, K tile staged with async global->LDS.
// ---------------------------------------------------------------------------
__global__ __launch_bounds__(128)
void k_flash_attn(const unsigned short* __restrict__ Qb,
                  const unsigned short* __restrict__ Kb,
                  const unsigned short* __restrict__ Vb,
                  unsigned short* __restrict__ Ob) {
    const float scale = 0.08838834764831845f;   // 1/sqrt(128)
    const int LK = HD + 8;
    const int LV = 32 + 8;
    __shared__ __align__(16) unsigned short Ks[32 * (HD + 8)];   // keys row-major
    __shared__ __align__(16) unsigned short Vt[HD * (32 + 8)];   // V transposed [hd][k]
    __shared__ __align__(16) unsigned short Ps[4][16 * 32];      // per-wave P patch

    const int tid  = threadIdx.x;
    const int lane = tid & 31;
    const int wave = tid >> 5;
    const int hlf  = lane >> 4;
    const int lr   = lane & 15;
    const int h    = blockIdx.x;        // query head
    const int g    = h >> 2;            // kv head (GQA 4:1)
    const int qb   = blockIdx.y * 64 + wave * 16;

    // Preload Q A-fragments for the 4 K-steps across HD=128 (stay in VGPRs)
    FragAB qf[4];
#pragma unroll
    for (int ks = 0; ks < 4; ++ks) {
        const unsigned short* qrow = &Qb[(size_t)(qb + lr) * QKD + (size_t)h * HD + ks * 32];
#pragma unroll
        for (int e = 0; e < 16; ++e) {
            int k = (e & 7) + ((e >> 3) << 4) + (hlf << 3);   // ISA 16-bit A layout
            qf[ks].u[e] = qrow[k];
        }
    }

    float mrow[8], lsum[8];
    v8f o[8];
    v8f zero = {};
#pragma unroll
    for (int r = 0; r < 8; ++r) { mrow[r] = -1e30f; lsum[r] = 0.f; }
#pragma unroll
    for (int j = 0; j < 8; ++j) o[j] = zero;

    for (int kb = 0; kb < KT; kb += 32) {
        // Async-stage K tile (32 x 128) row-major
        {
            int krow = tid >> 2, seg = (tid & 3) * 32;
            const unsigned short* src = &Kb[(size_t)(kb + krow) * KVD + (size_t)g * HD + seg];
            unsigned short* dst = &Ks[krow * LK + seg];
#pragma unroll
            for (int c = 0; c < 4; ++c)
                async_copy_b128(src + c * 8, dst + c * 8);
        }
        // Stage V tile transposed: Vt[hd][k] (needs lane shuffle, stays sync)
        {
            int krow = tid & 31, hdb = (tid >> 5) * 32;
            const uint4* src = (const uint4*)&Vb[(size_t)(kb + krow) * KVD + (size_t)g * HD + hdb];
            uint4 d[4]; d[0] = src[0]; d[1] = src[1]; d[2] = src[2]; d[3] = src[3];
            const unsigned short* ds = (const unsigned short*)d;
#pragma unroll
            for (int i = 0; i < 32; ++i)
                Vt[(hdb + i) * LV + krow] = ds[i];
        }
        if (kb + 32 < KT) {
            __builtin_prefetch(&Kb[(size_t)(kb + 32 + (tid >> 2)) * KVD + (size_t)g * HD], 0, 1);
            __builtin_prefetch(&Vb[(size_t)(kb + 32 + (tid & 31)) * KVD + (size_t)g * HD], 0, 1);
        }
        wait_async();
        __syncthreads();

        // S = Q * K^T  (two 16x16 score frags: keys 0..15 and 16..31 of block)
        v8f s[2]; s[0] = zero; s[1] = zero;
#pragma unroll
        for (int nf = 0; nf < 2; ++nf) {
#pragma unroll
            for (int ks = 0; ks < 4; ++ks) {
                FragAB kf;                          // B frag: N=key=lr, K=hd chunks (hlf*8),(16+hlf*8)
                int n = nf * 16 + lr;
                kf.q[0] = *(const uint4*)&Ks[n * LK + ks * 32 + hlf * 8];
                kf.q[1] = *(const uint4*)&Ks[n * LK + ks * 32 + 16 + hlf * 8];
                s[nf] = __builtin_amdgcn_wmma_f32_16x16x32_bf16(
                    false, qf[ks].v, false, kf.v, (short)0, s[nf], false, false);
            }
        }

        // Online softmax update (rows r / r+8 live in halves 0/1 of the wave)
#pragma unroll
        for (int r = 0; r < 8; ++r) {
            float s0 = s[0][r] * scale;
            float s1 = s[1][r] * scale;
            float t = half16_max(fmaxf(s0, s1));
            float mnew  = fmaxf(mrow[r], t);
            float alpha = __expf(mrow[r] - mnew);
            mrow[r] = mnew;
            float p0 = __expf(s0 - mnew);
            float p1 = __expf(s1 - mnew);
            float rs = half16_sum(p0 + p1);
            lsum[r] = lsum[r] * alpha + rs;
#pragma unroll
            for (int j = 0; j < 8; ++j) o[j][r] *= alpha;
            // C-layout -> LDS so we can re-read in A-layout
            Ps[wave][(r + hlf * 8) * 32 + lr]      = f32_to_bf16(p0);
            Ps[wave][(r + hlf * 8) * 32 + 16 + lr] = f32_to_bf16(p1);
        }

        // Re-read P as an A fragment (same wave; DS waits handle ordering)
        FragAB pf;
#pragma unroll
        for (int e = 0; e < 16; ++e) {
            int k = (e & 7) + ((e >> 3) << 4) + (hlf << 3);
            pf.u[e] = Ps[wave][lr * 32 + k];
        }

        // O += P * V  (8 hd frags, single K-step of 32 keys)
#pragma unroll
        for (int j = 0; j < 8; ++j) {
            FragAB vf;                              // B frag: N=hd=j*16+lr, K chunks (hlf*8),(16+hlf*8)
            vf.q[0] = *(const uint4*)&Vt[(j * 16 + lr) * LV + hlf * 8];
            vf.q[1] = *(const uint4*)&Vt[(j * 16 + lr) * LV + 16 + hlf * 8];
            o[j] = __builtin_amdgcn_wmma_f32_16x16x32_bf16(
                false, pf.v, false, vf.v, (short)0, o[j], false, false);
        }
        __syncthreads();
    }

    // Normalize and write attention output as bf16 (q_len, NH*HD)
#pragma unroll
    for (int j = 0; j < 8; ++j) {
#pragma unroll
        for (int r = 0; r < 8; ++r) {
            int qg = qb + r + hlf * 8;
            float v = o[j][r] / lsum[r];
            Ob[(size_t)qg * QKD + (size_t)h * HD + j * 16 + lr] = f32_to_bf16(v);
        }
    }
}

// ---------------------------------------------------------------------------
// Host-side orchestration
// ---------------------------------------------------------------------------
extern "C" void kernel_launch(void* const* d_in, const int* in_sizes, int n_in,
                              void* d_out, int out_size, void* d_ws, size_t ws_size,
                              hipStream_t stream) {
    (void)in_sizes; (void)n_in; (void)out_size; (void)ws_size;
    const float* hs   = (const float*)d_in[0];   // (QL, H)
    const float* th   = (const float*)d_in[1];   // (CTX, H)
    const float* cosb = (const float*)d_in[2];   // (CTX+QL, HD)
    const float* sinb = (const float*)d_in[3];
    const float* Wq   = (const float*)d_in[4];   // (NH*HD, H)
    const float* Wk   = (const float*)d_in[5];   // (NKV*HD, H)
    const float* Wv   = (const float*)d_in[6];
    const float* Wo   = (const float*)d_in[7];   // (H, NH*HD)
    float* out = (float*)d_out;

    char* wsp = (char*)d_ws;
    auto alloc = [&](size_t elems) {
        char* p = wsp;
        wsp += ((elems * 2) + 255) & ~(size_t)255;
        return (unsigned short*)p;
    };
    unsigned short* hb  = alloc((size_t)QL  * HDIM);
    unsigned short* tb  = alloc((size_t)CTXL* HDIM);
    unsigned short* wqb = alloc((size_t)QKD * HDIM);
    unsigned short* wkb = alloc((size_t)KVD * HDIM);
    unsigned short* wvb = alloc((size_t)KVD * HDIM);
    unsigned short* wob = alloc((size_t)HDIM* QKD);
    unsigned short* Qf  = alloc((size_t)QL  * QKD);
    unsigned short* Kf  = alloc((size_t)KT  * KVD);
    unsigned short* Vf  = alloc((size_t)KT  * KVD);
    unsigned short* Ab  = alloc((size_t)QL  * QKD);

    const int CVT_T = 256, CVT_B = 2048;
    k_f32_to_bf16<<<CVT_B, CVT_T, 0, stream>>>(hs, hb, QL * HDIM);
    k_f32_to_bf16<<<CVT_B, CVT_T, 0, stream>>>(th, tb, CTXL * HDIM);
    k_f32_to_bf16<<<CVT_B, CVT_T, 0, stream>>>(Wq, wqb, QKD * HDIM);
    k_f32_to_bf16<<<CVT_B, CVT_T, 0, stream>>>(Wk, wkb, KVD * HDIM);
    k_f32_to_bf16<<<CVT_B, CVT_T, 0, stream>>>(Wv, wvb, KVD * HDIM);
    k_f32_to_bf16<<<CVT_B, CVT_T, 0, stream>>>(Wo, wob, HDIM * QKD);

    // Projections (C[M,N] = A[M,K] * W[N,K]^T); K/V write into concat buffers
    dim3 gq(QKD / 128, QL / 128);
    dim3 gkv(KVD / 128, QL / 128);
    k_gemm_bf16<unsigned short><<<gq,  256, 0, stream>>>(hb, wqb, Qf, QL, QKD, HDIM);
    k_gemm_bf16<unsigned short><<<gkv, 256, 0, stream>>>(tb, wkb, Kf,                      CTXL, KVD, HDIM);
    k_gemm_bf16<unsigned short><<<gkv, 256, 0, stream>>>(hb, wkb, Kf + (size_t)CTXL * KVD, QL,   KVD, HDIM);
    k_gemm_bf16<unsigned short><<<gkv, 256, 0, stream>>>(tb, wvb, Vf,                      CTXL, KVD, HDIM);
    k_gemm_bf16<unsigned short><<<gkv, 256, 0, stream>>>(hb, wvb, Vf + (size_t)CTXL * KVD, QL,   KVD, HDIM);

    // RoPE: Q uses positions CTX..CTX+QL-1; K rows align with positions 0..KT-1
    {
        int totq = QL * NH * (HD / 2);
        k_rope<<<(totq + 255) / 256, 256, 0, stream>>>(Qf, cosb + (size_t)CTXL * HD,
                                                       sinb + (size_t)CTXL * HD, QL, NH);
        int totk = KT * NKV * (HD / 2);
        k_rope<<<(totk + 255) / 256, 256, 0, stream>>>(Kf, cosb, sinb, KT, NKV);
    }

    // Flash attention: grid (heads, q-blocks of 64), 128 threads
    dim3 ga(NH, QL / 64);
    k_flash_attn<<<ga, 128, 0, stream>>>(Qf, Kf, Vf, Ab);

    // Output projection -> fp32 d_out
    dim3 go(HDIM / 128, QL / 128);
    k_gemm_bf16<float><<<go, 256, 0, stream>>>(Ab, wob, out, QL, HDIM, QKD);
}
